// Net_2405181686362
// MI455X (gfx1250) — compile-verified, matrix operand order checked by reading, not run on hardware
//
#include <hip/hip_runtime.h>
#include <hip/hip_bf16.h>
#include <math.h>

typedef __attribute__((ext_vector_type(16))) _Float16 v16h;
typedef __attribute__((ext_vector_type(8)))  float    v8f;

#define HDIM 32

// ---------------------------------------------------------------------------
// Kernel A: per-row feature transform.
//   xw1t  = (f16) transpose(x@w1)  laid out [B, H, Nn]  (WMMA B-operand feed)
//   xw2b  = x@w2 + bias            laid out [B, Nn, H]  (epilogue add)
// ---------------------------------------------------------------------------
__global__ void k_xw(const float* __restrict__ x,
                     const float* __restrict__ w1,
                     const float* __restrict__ w2,
                     const float* __restrict__ bias,
                     int K, int Nn, long rows,
                     _Float16* __restrict__ xw1t,
                     float* __restrict__ xw2b) {
  long gid = (long)blockIdx.x * blockDim.x + threadIdx.x;
  if (gid >= rows * HDIM) return;
  int  h   = (int)(gid & (HDIM - 1));
  long row = gid >> 5;                       // global node row = b*Nn + n
  const float* xr = x + row * (long)K;
  float a1 = 0.f, a2 = 0.f;
  for (int f = 0; f < K; ++f) {
    float xv = xr[f];
    a1 = fmaf(xv, w1[f * HDIM + h], a1);
    a2 = fmaf(xv, w2[f * HDIM + h], a2);
  }
  long bb = row / Nn;
  long nn = row - bb * Nn;
  xw1t[(bb * HDIM + h) * Nn + nn] = (_Float16)a1;
  xw2b[gid]                       = a2 + bias[h];
}

// ---------------------------------------------------------------------------
// Kernel B: Z = relu( Agath @ XW1 + XW2b ), batched, WMMA f16 -> f32 acc.
//   Agath[m,j] = GATHER ? A[b, idx[b,m], idx[b,j]] : A[b, m0+m, j0+j]
// Block: 64 threads (2 waves); wave w owns output columns [16w,16w+16).
// K stepped by 64 => two v_wmma per stage/sync pair.
// sA[m][k] and sBt[n][k] both use the ISA 7.12.2 K-pair packing so both
// fragment reads merge into ds_load_2addr_b64.
// ---------------------------------------------------------------------------
template <bool GATHER>
__global__ __launch_bounds__(64)
void k_gcs_wmma(const float* __restrict__ A,
                const int*   __restrict__ idx,     // used when GATHER
                int Nsrc,                          // node dim of A (512)
                const _Float16* __restrict__ XW1t, // [B,H,Nn] f16 transposed
                const float*    __restrict__ XW2b, // [B,Nn,H] f32
                float* __restrict__ Z,             // [B,Nn,H] f32
                int Nn) {
  __shared__ __align__(16) _Float16 sA[16][80];   // m x k tile (16B-mult rows)
  __shared__ __align__(16) _Float16 sBt[32][80];  // n x k tile (transposed B)
  __shared__ int sRow[16];
  __shared__ int sCol[64];

  const int b    = blockIdx.y;
  const int m0   = blockIdx.x * 16;
  const int t    = threadIdx.x;
  const int w    = t >> 5;        // wave id -> column half
  const int lane = t & 31;
  const int half = lane >> 4;
  const int l16  = lane & 15;

  const size_t    Abase = (size_t)b * Nsrc * Nsrc;
  const _Float16* Bsrc  = XW1t + (size_t)b * HDIM * Nn;
  const float*    XW2bb = XW2b + (size_t)b * Nn * HDIM;

  if (GATHER && t < 16) sRow[t] = idx[(size_t)b * Nn + m0 + t];

  v8f c = {};

  for (int j0 = 0; j0 < Nn; j0 += 64) {
    __syncthreads();                 // prev fragments consumed; sRow visible
    if (GATHER) {
      sCol[t] = idx[(size_t)b * Nn + j0 + t];
      __syncthreads();
#pragma unroll
      for (int i = 0; i < 16; ++i) {            // 1024 gathered elements
        int l = t * 16 + i;
        int r = l >> 6, cc = l & 63;
        sA[r][cc] = (_Float16)A[Abase + (size_t)sRow[r] * Nsrc + sCol[cc]];
      }
    } else {
#pragma unroll
      for (int i = 0; i < 4; ++i) {             // 256 float4 loads (16x64)
        int q  = t * 4 + i;
        int r  = q >> 4;
        int cc = (q & 15) * 4;
        const float4 v =
            *(const float4*)(A + Abase + (size_t)(m0 + r) * Nsrc + j0 + cc);
        sA[r][cc + 0] = (_Float16)v.x;
        sA[r][cc + 1] = (_Float16)v.y;
        sA[r][cc + 2] = (_Float16)v.z;
        sA[r][cc + 3] = (_Float16)v.w;
      }
      if (j0 + 64 < Nn) {  // stream-prefetch next A tile row
        __builtin_prefetch(
            A + Abase + (size_t)(m0 + (t & 15)) * Nsrc + j0 + 64, 0, 1);
      }
    }
    // B tile: pure 16B-vector copy of pre-converted f16 transposed XW1.
    // 32 rows (n) x 64 halves (k) = 4KB = 256 chunks of 16B.
#pragma unroll
    for (int i = 0; i < 4; ++i) {
      int q  = t * 4 + i;
      int r  = q >> 3;            // 8 chunks per row
      int ch = (q & 7) * 8;       // half offset, 16B aligned
      const float4 v = *(const float4*)(Bsrc + (size_t)r * Nn + j0 + ch);
      *(float4*)&sBt[r][ch] = v;
    }
    __syncthreads();

    // Two WMMAs: K chunks [j0, j0+32) and [j0+32, j0+64).
    // Fragment packing (ISA 7.12.2): VGPR v holds K = {2v,2v+1} (+16 if v>=4),
    // shifted by 8*half.
#pragma unroll
    for (int u = 0; u < 2; ++u) {
      union { v16h v; _Float16 f[16]; } ua, ub;
#pragma unroll
      for (int e = 0; e < 16; ++e) {
        int vv = e >> 1;
        int kk = ((vv < 4) ? (2 * vv) : (16 + 2 * (vv - 4)))
                 + 8 * half + (e & 1) + 32 * u;
        ua.f[e] = sA[l16][kk];              // A row m = lane%16
        ub.f[e] = sBt[w * 16 + l16][kk];    // B col n = lane%16 (+wave off)
      }
      c = __builtin_amdgcn_wmma_f32_16x16x32_f16(
          /*neg_a=*/false, ua.v, /*neg_b=*/false, ub.v,
          /*c_mod=*/(short)0, c, /*reuse_a=*/false, /*reuse_b=*/false);
    }
  }

  // Epilogue: C/D layout — VGPR r: M = r + 8*half, N = lane%16 (+16w)
  union { v8f v; float f[8]; } uc; uc.v = c;
#pragma unroll
  for (int r = 0; r < 8; ++r) {
    int m = r + 8 * half;
    int n = w * 16 + l16;
    size_t o   = (size_t)(m0 + m) * HDIM + n;
    float  val = uc.f[r] + XW2bb[o];
    Z[(size_t)b * Nn * HDIM + o] = val > 0.f ? val : 0.f;
  }
}

// ---------------------------------------------------------------------------
// Kernel C: exact top-k (jax.lax.top_k semantics: descending, stable ties).
// One block per graph, blockDim = N. Writes composed (global) indices so the
// adjacency slice is never materialized, plus the gated kept features.
// ---------------------------------------------------------------------------
__global__ void k_topk(const float* __restrict__ X,      // [B,N,H]
                       const float* __restrict__ p,      // [H]
                       const int*   __restrict__ parent, // [B,N] or null
                       int N, int k,
                       int*   __restrict__ idx_out,      // [B,k] global ids
                       float* __restrict__ xk) {         // [B,k,H]
  extern __shared__ float ys[];
  int b = blockIdx.x;
  int n = threadIdx.x;

  float ss = 0.f;
#pragma unroll
  for (int h = 0; h < HDIM; ++h) ss = fmaf(p[h], p[h], ss);
  float inv = rsqrtf(ss);

  const float* xr = X + ((size_t)b * N + n) * HDIM;
  float y = 0.f;
#pragma unroll
  for (int h = 0; h < HDIM; ++h) y = fmaf(xr[h], p[h], y);
  y *= inv;
  ys[n] = y;
  __syncthreads();

  int rank = 0;
  for (int j = 0; j < N; ++j) {
    float yj = ys[j];
    rank += (yj > y) || (yj == y && j < n);
  }
  if (rank < k) {
    idx_out[(size_t)b * k + rank] = parent ? parent[(size_t)b * N + n] : n;
    float g = 1.f / (1.f + __expf(-y));
    float* dst = xk + ((size_t)b * k + rank) * HDIM;
#pragma unroll
    for (int h = 0; h < HDIM; ++h) dst[h] = xr[h] * g;
  }
}

// ---------------------------------------------------------------------------
// Kernel E: mean pool over nodes + dense(32->16) + softmax. 1 wave per graph.
// ---------------------------------------------------------------------------
__global__ __launch_bounds__(32)
void k_head(const float* __restrict__ X, int N,          // [B,N,32]
            const float* __restrict__ wd,                // [32,16]
            const float* __restrict__ bd,                // [16]
            float* __restrict__ out) {                   // [B,16]
  __shared__ float pooled[HDIM];
  __shared__ float lg[16];
  int b = blockIdx.x;
  int h = threadIdx.x;

  float s = 0.f;
  const float* xb = X + (size_t)b * N * HDIM;
  for (int n = 0; n < N; ++n) s += xb[(size_t)n * HDIM + h];
  pooled[h] = s / (float)N;
  __syncthreads();

  if (h < 16) {
    float acc = bd[h];
#pragma unroll
    for (int q = 0; q < HDIM; ++q) acc = fmaf(pooled[q], wd[q * 16 + h], acc);
    lg[h] = acc;
  }
  __syncthreads();
  if (h < 16) {
    float mx = lg[0];
#pragma unroll
    for (int q = 1; q < 16; ++q) mx = fmaxf(mx, lg[q]);
    float se = 0.f;
#pragma unroll
    for (int q = 0; q < 16; ++q) se += __expf(lg[q] - mx);
    out[(size_t)b * 16 + h] = __expf(lg[h] - mx) / se;
  }
}

// ---------------------------------------------------------------------------
extern "C" void kernel_launch(void* const* d_in, const int* in_sizes, int n_in,
                              void* d_out, int out_size, void* d_ws, size_t ws_size,
                              hipStream_t stream) {
  (void)in_sizes; (void)n_in; (void)out_size; (void)ws_size;
  const int B = 256, N0 = 512, F0 = 128, K1 = 256, K2 = 128;

  const float* x      = (const float*)d_in[0];
  const float* a      = (const float*)d_in[1];
  const float* w1_1   = (const float*)d_in[2];
  const float* w2_1   = (const float*)d_in[3];
  const float* b1     = (const float*)d_in[4];
  const float* w1_2   = (const float*)d_in[5];
  const float* w2_2   = (const float*)d_in[6];
  const float* b2     = (const float*)d_in[7];
  const float* w1_3   = (const float*)d_in[8];
  const float* w2_3   = (const float*)d_in[9];
  const float* b3     = (const float*)d_in[10];
  const float* p_pool = (const float*)d_in[11];
  const float* wd     = (const float*)d_in[12];
  const float* bd     = (const float*)d_in[13];

  // Workspace carve (buffers reused across layers; ~51 MB total)
  char* ws = (char*)d_ws;
  size_t off = 0;
  auto carve = [&](size_t bytes) -> void* {
    void* p = ws + off;
    off += (bytes + 255) & ~(size_t)255;
    return p;
  };
  _Float16* xw1t = (_Float16*)carve((size_t)B * N0 * HDIM * sizeof(_Float16));
  float*    xw2b = (float*)   carve((size_t)B * N0 * HDIM * sizeof(float));
  float*    Zbuf = (float*)   carve((size_t)B * N0 * HDIM * sizeof(float));
  float*    xg   = (float*)   carve((size_t)B * K1 * HDIM * sizeof(float));
  int*      idx1 = (int*)     carve((size_t)B * K1 * sizeof(int));
  int*      idx2 = (int*)     carve((size_t)B * K2 * sizeof(int));

  const int THR = 256;

  // ---- Layer 1: GCSConv on full graph (direct, streaming A) ----
  long rows1 = (long)B * N0;
  k_xw<<<dim3((unsigned)((rows1 * HDIM + THR - 1) / THR)), THR, 0, stream>>>(
      x, w1_1, w2_1, b1, F0, N0, rows1, xw1t, xw2b);
  k_gcs_wmma<false><<<dim3(N0 / 16, B), 64, 0, stream>>>(
      a, nullptr, N0, xw1t, xw2b, Zbuf, N0);

  // ---- Pool 1: 512 -> 256 (indices are global node ids) ----
  k_topk<<<B, N0, N0 * sizeof(float), stream>>>(
      Zbuf, p_pool, nullptr, N0, K1, idx1, xg);

  // ---- Layer 2: GCSConv; adjacency gathered through idx1 (no a1 buffer) ----
  long rows2 = (long)B * K1;
  k_xw<<<dim3((unsigned)((rows2 * HDIM + THR - 1) / THR)), THR, 0, stream>>>(
      xg, w1_2, w2_2, b2, HDIM, K1, rows2, xw1t, xw2b);
  k_gcs_wmma<true><<<dim3(K1 / 16, B), 64, 0, stream>>>(
      a, idx1, N0, xw1t, xw2b, Zbuf, K1);

  // ---- Pool 2: 256 -> 128; compose indices back to the original graph ----
  k_topk<<<B, K1, K1 * sizeof(float), stream>>>(
      Zbuf, p_pool, idx1, K1, K2, idx2, xg);

  // ---- Layer 3: GCSConv; adjacency gathered through composed idx2 ----
  long rows3 = (long)B * K2;
  k_xw<<<dim3((unsigned)((rows3 * HDIM + THR - 1) / THR)), THR, 0, stream>>>(
      xg, w1_3, w2_3, b3, HDIM, K2, rows3, xw1t, xw2b);
  k_gcs_wmma<true><<<dim3(K2 / 16, B), 64, 0, stream>>>(
      a, idx2, N0, xw1t, xw2b, Zbuf, K2);

  // ---- Head: mean pool + dense + softmax ----
  k_head<<<B, 32, 0, stream>>>(Zbuf, K2, wd, bd, (float*)d_out);
}
